// LightweightMambaBlock_28106265985453
// MI455X (gfx1250) — matrix operand order repaired and checked by reading, not compiled, not run on hardware
//
#include <hip/hip_runtime.h>
#include <hip/hip_bf16.h>
#include <math.h>

typedef __attribute__((ext_vector_type(16))) __bf16 v16bf;
typedef __attribute__((ext_vector_type(8)))  __bf16 v8bf;
typedef __attribute__((ext_vector_type(8)))  float  v8f;

#define D_MODEL 192
#define D_INNER 230
#define D_STATE 16
#define DT_RANK 15
#define XDBL_W  48          // 47 padded to 48 (3 n-tiles)
#define KPAD    256         // padded inner dim (8 chunks of 32)
#define BATCH   4
#define SEQ     4096        // 64*64
#define MROWS   (BATCH*SEQ) // 16384
#define MTILES  (MROWS/16)  // 1024
#define LN_EPS  1e-5f

// ---------------------------------------------------------------------------
// Transpose+convert x (NCHW f32) -> xbf [m = b*SEQ+l][c] bf16, LDS-tiled.
// ---------------------------------------------------------------------------
__global__ void k_xpose(const float* __restrict__ x, __bf16* __restrict__ xbf) {
    __shared__ float t[32][33];
    const int b = blockIdx.z;
    const int l0 = blockIdx.x * 32, c0 = blockIdx.y * 32;
#pragma unroll
    for (int j = 0; j < 32; j += 8)
        t[threadIdx.y + j][threadIdx.x] =
            x[((size_t)b * D_MODEL + c0 + threadIdx.y + j) * SEQ + l0 + threadIdx.x];
    __syncthreads();
#pragma unroll
    for (int j = 0; j < 32; j += 8)
        xbf[((size_t)b * SEQ + l0 + threadIdx.y + j) * D_MODEL + c0 + threadIdx.x] =
            (__bf16)t[threadIdx.x][threadIdx.y + j];
}

// ---------------------------------------------------------------------------
// Pre-swizzle a weight matrix [K][N] (f32) into WMMA-B lane order, zero-padded:
// dst[((nt*KT + kc)*32 + lane)*16 + i] = W[kc*32 + (lane>>4)*16 + i][nt*16 + (lane&15)]
// -> in the GEMM each lane reads its 16 bf16 as two contiguous b128 loads.
// ---------------------------------------------------------------------------
__global__ void k_swzB(const float* __restrict__ src, __bf16* __restrict__ dst,
                       int srcK, int srcN, int KT, int NT) {
    int i = blockIdx.x * blockDim.x + threadIdx.x;
    int total = NT * KT * 32 * 16;
    if (i >= total) return;
    int ii = i & 15;
    int lane = (i >> 4) & 31;
    int kc = (i >> 9) % KT;
    int nt = (i >> 9) / KT;
    int k = kc * 32 + (lane >> 4) * 16 + ii;
    int n = nt * 16 + (lane & 15);
    float v = (k < srcK && n < srcN) ? src[(size_t)k * srcN + n] : 0.0f;
    dst[i] = (__bf16)v;
}

// ---------------------------------------------------------------------------
// Register-blocked WMMA core: one m-tile row fragment feeds NTW accumulators.
// A: row-major bf16 [*, ldA] (ldA multiple of 16B); Bsw: swizzled (above).
// ---------------------------------------------------------------------------
template <int KT, int NTW>
__device__ __forceinline__ void wmma_rowtiles(const __bf16* __restrict__ A,
                                              const __bf16* __restrict__ Bsw,
                                              int m0, int nt0, int ldA, int lane,
                                              v8f* acc) {
    const int sub = lane >> 4, mrow = lane & 15;
    const __bf16* arow = A + (size_t)(m0 + mrow) * ldA;
#pragma unroll
    for (int kc = 0; kc < KT; ++kc) {
        v8bf alo = *(const v8bf*)(arow + kc * 32 + sub * 16);
        v8bf ahi = *(const v8bf*)(arow + kc * 32 + sub * 16 + 8);
        v16bf a = __builtin_shufflevector(alo, ahi, 0, 1, 2, 3, 4, 5, 6, 7,
                                          8, 9, 10, 11, 12, 13, 14, 15);
#pragma unroll
        for (int t = 0; t < NTW; ++t) {
            const v8bf* bp =
                (const v8bf*)(Bsw + ((((size_t)(nt0 + t) * KT + kc) * 32 + lane) << 4));
            v16bf b = __builtin_shufflevector(bp[0], bp[1], 0, 1, 2, 3, 4, 5, 6, 7,
                                              8, 9, 10, 11, 12, 13, 14, 15);
            acc[t] = __builtin_amdgcn_wmma_f32_16x16x32_bf16(false, a, false, b,
                                                             (short)0, acc[t],
                                                             false, false);
        }
    }
}

// ---------------------------------------------------------------------------
// GEMM1: u = silu(xbf @ W_in)  K=192 (KT=6), 16 n-tiles, 4 per wave.
// ---------------------------------------------------------------------------
__global__ void k_gemm1(const __bf16* __restrict__ xbf, const __bf16* __restrict__ Bsw,
                        float* __restrict__ u_f32, __bf16* __restrict__ u_bf) {
    const int lane = threadIdx.x & 31, wv = threadIdx.x >> 5;
    const int mtile = blockIdx.x;
    const int nt0 = wv * 4;
    v8f acc[4] = {};
    wmma_rowtiles<6, 4>(xbf, Bsw, mtile * 16, nt0, D_MODEL, lane, acc);
    const int sub = lane >> 4, mrow = lane & 15;
#pragma unroll
    for (int t = 0; t < 4; ++t) {
        int d = (nt0 + t) * 16 + mrow;
#pragma unroll
        for (int r = 0; r < 8; ++r) {
            int mm = mtile * 16 + sub * 8 + r;
            float v = acc[t][r];
            float s = v * __builtin_amdgcn_rcpf(1.0f + __expf(-v)); // fast silu
            u_bf[(size_t)mm * KPAD + d] = (__bf16)s;                // pad cols -> 0
            if (d < D_INNER) u_f32[(size_t)mm * D_INNER + d] = s;
        }
    }
}

// ---------------------------------------------------------------------------
// GEMM2: x_dbl = u @ x_proj_w  K=KPAD (KT=8), 3 n-tiles, all in one wave.
// ---------------------------------------------------------------------------
__global__ void k_gemm2(const __bf16* __restrict__ u_bf, const __bf16* __restrict__ Bsw,
                        float* __restrict__ xdbl) {
    const int lane = threadIdx.x & 31, wv = threadIdx.x >> 5;
    const int mtile = blockIdx.x * 4 + wv;
    v8f acc[3] = {};
    wmma_rowtiles<8, 3>(u_bf, Bsw, mtile * 16, 0, KPAD, lane, acc);
    const int sub = lane >> 4, mrow = lane & 15;
#pragma unroll
    for (int t = 0; t < 3; ++t)
#pragma unroll
        for (int r = 0; r < 8; ++r)
            xdbl[(size_t)(mtile * 16 + sub * 8 + r) * XDBL_W + t * 16 + mrow] = acc[t][r];
}

// ---------------------------------------------------------------------------
// GEMM3: o = y_ln @ W_out  K=KPAD (KT=8), 12 n-tiles, 6 per wave (48 WMMAs).
// ---------------------------------------------------------------------------
__global__ void k_gemm3(const __bf16* __restrict__ yln, const __bf16* __restrict__ Bsw,
                        float* __restrict__ o) {
    const int lane = threadIdx.x & 31, wv = threadIdx.x >> 5;
    const int mtile = blockIdx.x * 2 + (wv >> 1);
    const int nt0 = (wv & 1) * 6;
    v8f acc[6] = {};
    wmma_rowtiles<8, 6>(yln, Bsw, mtile * 16, nt0, KPAD, lane, acc);
    const int sub = lane >> 4, mrow = lane & 15;
#pragma unroll
    for (int t = 0; t < 6; ++t)
#pragma unroll
        for (int r = 0; r < 8; ++r)
            o[(size_t)(mtile * 16 + sub * 8 + r) * D_MODEL + (nt0 + t) * 16 + mrow] =
                acc[t][r];
}

// ---------------------------------------------------------------------------
// dt = softplus(dt_r @ dt_proj_w + dt_proj_b)   (K=15: plain VALU)
// ---------------------------------------------------------------------------
__global__ void k_dt(const float* __restrict__ xdbl, const float* __restrict__ dtw,
                     const float* __restrict__ dtb, float* __restrict__ dt) {
    size_t i = (size_t)blockIdx.x * blockDim.x + threadIdx.x;
    if (i >= (size_t)MROWS * D_INNER) return;
    size_t m = i / D_INNER;
    int d = (int)(i % D_INNER);
    float s = dtb[d];
#pragma unroll
    for (int r = 0; r < DT_RANK; ++r)
        s = fmaf(xdbl[m * XDBL_W + r], dtw[(size_t)r * D_INNER + d], s);
    dt[i] = (s > 20.0f) ? s : log1pf(__expf(s));
}

// ---------------------------------------------------------------------------
// Selective scan: one block per batch, one thread per d, h[16] in VGPRs.
// Per-step B_t/C_t broadcast fetched with gfx1250 async global->LDS DMA
// (ASYNCcnt path), issued by wave 0 into dynamic LDS at offset 0.
// ---------------------------------------------------------------------------
__global__ void k_scan(const float* __restrict__ u, const float* __restrict__ dt,
                       const float* __restrict__ xdbl, const float* __restrict__ A_log,
                       const float* __restrict__ Dvec, float* __restrict__ y) {
    extern __shared__ float smem[];   // [0..15]=B_t, [16..31]=C_t (dyn LDS @ 0)
    const int b = blockIdx.x;
    const int d = threadIdx.x;
    const int lane = threadIdx.x & 31;
    const bool w0 = (threadIdx.x >> 5) == 0;
    const bool active = d < D_INNER;
    float h[D_STATE], Ad[D_STATE];
    float Dd = 0.0f;
    if (active) {
#pragma unroll
        for (int n = 0; n < D_STATE; ++n) {
            Ad[n] = -__expf(A_log[(size_t)d * D_STATE + n]);
            h[n] = 0.0f;
        }
        Dd = Dvec[d];
    }
    const unsigned int ldsb = lane * 4u;
    for (int l = 0; l < SEQ; ++l) {
        size_t row = (size_t)b * SEQ + l;
        if (w0) {
            unsigned long long ga =
                (unsigned long long)(xdbl + row * XDBL_W + DT_RANK + lane);
            asm volatile("global_load_async_to_lds_b32 %0, %1, off"
                         :: "v"(ldsb), "v"(ga) : "memory");
            asm volatile("s_wait_asynccnt 0x0" ::: "memory");
        }
        __syncthreads();
        if (active) {
            if (l + 8 < SEQ) {  // lookahead into L2 for the serial chain
                __builtin_prefetch(&dt[(row + 8) * D_INNER + d], 0, 1);
                __builtin_prefetch(&u [(row + 8) * D_INNER + d], 0, 1);
            }
            float dtv = dt[row * D_INNER + d];
            float uv  = u [row * D_INNER + d];
            float du  = dtv * uv;
            float yv  = 0.0f;
#pragma unroll
            for (int n = 0; n < D_STATE; ++n) {
                float dA = __expf(dtv * Ad[n]);
                h[n] = fmaf(dA, h[n], du * smem[n]);
                yv   = fmaf(h[n], smem[D_STATE + n], yv);
            }
            y[row * D_INNER + d] = fmaf(uv, Dd, yv);
        }
        __syncthreads();
    }
}

// ---------------------------------------------------------------------------
// LayerNorm over D_INNER per row; one wave32 per row; writes padded bf16.
// ---------------------------------------------------------------------------
__global__ void k_ln(const float* __restrict__ y, const float* __restrict__ g,
                     const float* __restrict__ bb, __bf16* __restrict__ yout) {
    const int lane = threadIdx.x & 31;
    const int row = blockIdx.x * 8 + (threadIdx.x >> 5);
    const float* yr = y + (size_t)row * D_INNER;
    float s = 0.0f, s2 = 0.0f;
    for (int d = lane; d < D_INNER; d += 32) {
        float v = yr[d];
        s += v; s2 += v * v;
    }
#pragma unroll
    for (int off = 16; off; off >>= 1) {
        s  += __shfl_xor(s,  off, 32);
        s2 += __shfl_xor(s2, off, 32);
    }
    float mu  = s / (float)D_INNER;
    float var = s2 / (float)D_INNER - mu * mu;
    float inv = rsqrtf(var + LN_EPS);
    __bf16* orow = yout + (size_t)row * KPAD;
    for (int d = lane; d < KPAD; d += 32) {
        float v = (d < D_INNER) ? fmaf((yr[d] - mu) * inv, g[d], bb[d]) : 0.0f;
        orow[d] = (__bf16)v;
    }
}

// ---------------------------------------------------------------------------
// BatchNorm stats per channel (over B*H*W = 16384 samples)
// ---------------------------------------------------------------------------
__global__ void k_bnstats(const float* __restrict__ o, float* __restrict__ stats) {
    const int c = blockIdx.x;
    float s = 0.0f, s2 = 0.0f;
    for (int i = threadIdx.x; i < MROWS; i += blockDim.x) {
        float v = o[(size_t)i * D_MODEL + c];
        s += v; s2 += v * v;
    }
    __shared__ float rs[256], rs2[256];
    rs[threadIdx.x] = s; rs2[threadIdx.x] = s2;
    __syncthreads();
    for (int st = 128; st; st >>= 1) {
        if (threadIdx.x < st) {
            rs[threadIdx.x]  += rs[threadIdx.x + st];
            rs2[threadIdx.x] += rs2[threadIdx.x + st];
        }
        __syncthreads();
    }
    if (threadIdx.x == 0) {
        float mu = rs[0] / (float)MROWS;
        stats[c] = mu;
        stats[D_MODEL + c] = rs2[0] / (float)MROWS - mu * mu;
    }
}

// ---------------------------------------------------------------------------
// BatchNorm apply + transpose to NCHW output layout
// ---------------------------------------------------------------------------
__global__ void k_bnapply(const float* __restrict__ o, const float* __restrict__ stats,
                          const float* __restrict__ g, const float* __restrict__ bb,
                          float* __restrict__ out) {
    size_t i = (size_t)blockIdx.x * blockDim.x + threadIdx.x;  // NCHW index
    if (i >= (size_t)BATCH * D_MODEL * SEQ) return;
    int l = (int)(i & (SEQ - 1));
    int c = (int)((i >> 12) % D_MODEL);
    int b = (int)(i / ((size_t)D_MODEL * SEQ));
    float mu = stats[c], var = stats[D_MODEL + c];
    float v = o[((size_t)b * SEQ + l) * D_MODEL + c];
    out[i] = fmaf((v - mu) * rsqrtf(var + LN_EPS), g[c], bb[c]);
}

// ---------------------------------------------------------------------------
extern "C" void kernel_launch(void* const* d_in, const int* in_sizes, int n_in,
                              void* d_out, int out_size, void* d_ws, size_t ws_size,
                              hipStream_t stream) {
    (void)in_sizes; (void)n_in; (void)out_size; (void)ws_size;
    const float* x         = (const float*)d_in[0];
    const float* W_in      = (const float*)d_in[1];
    const float* x_proj_w  = (const float*)d_in[2];
    const float* dt_proj_w = (const float*)d_in[3];
    const float* dt_proj_b = (const float*)d_in[4];
    const float* A_log     = (const float*)d_in[5];
    const float* Dvec      = (const float*)d_in[6];
    const float* ln_g      = (const float*)d_in[7];
    const float* ln_b      = (const float*)d_in[8];
    const float* W_out     = (const float*)d_in[9];
    const float* bn_g      = (const float*)d_in[10];
    const float* bn_b      = (const float*)d_in[11];
    float* out = (float*)d_out;

    char* w = (char*)d_ws;
    auto alloc = [&](size_t bytes) {
        void* p = (void*)w;
        w += (bytes + 255) & ~(size_t)255;
        return p;
    };
    float*  u_f32  = (float*)alloc((size_t)MROWS * D_INNER * 4);
    float*  dt     = (float*)alloc((size_t)MROWS * D_INNER * 4);
    float*  y      = (float*)alloc((size_t)MROWS * D_INNER * 4);
    float*  xdbl   = (float*)alloc((size_t)MROWS * XDBL_W * 4);
    float*  o_buf  = (float*)alloc((size_t)MROWS * D_MODEL * 4);
    __bf16* xbf    = (__bf16*)alloc((size_t)MROWS * D_MODEL * 2);
    __bf16* u_bf   = (__bf16*)alloc((size_t)MROWS * KPAD * 2);
    __bf16* yln_bf = (__bf16*)alloc((size_t)MROWS * KPAD * 2);
    __bf16* Bsw1   = (__bf16*)alloc((size_t)16 * 6 * 32 * 16 * 2);
    __bf16* Bsw2   = (__bf16*)alloc((size_t)3  * 8 * 32 * 16 * 2);
    __bf16* Bsw3   = (__bf16*)alloc((size_t)12 * 8 * 32 * 16 * 2);
    float*  stats  = (float*)alloc(2 * D_MODEL * 4);

    // input transpose + weight swizzles
    k_xpose<<<dim3(SEQ / 32, D_MODEL / 32, BATCH), dim3(32, 8), 0, stream>>>(x, xbf);
    k_swzB<<<(16 * 6 * 512 + 255) / 256, 256, 0, stream>>>(W_in, Bsw1, D_MODEL, D_INNER, 6, 16);
    k_swzB<<<(3 * 8 * 512 + 255) / 256, 256, 0, stream>>>(x_proj_w, Bsw2, D_INNER,
                                                          DT_RANK + 2 * D_STATE, 8, 3);
    k_swzB<<<(12 * 8 * 512 + 255) / 256, 256, 0, stream>>>(W_out, Bsw3, D_INNER, D_MODEL, 8, 12);

    // GEMM1: u = silu(x^T W_in)
    k_gemm1<<<MTILES, 128, 0, stream>>>(xbf, Bsw1, u_f32, u_bf);
    // GEMM2: x_dbl = u x_proj_w
    k_gemm2<<<MTILES / 4, 128, 0, stream>>>(u_bf, Bsw2, xdbl);
    // dt = softplus(dt_r dt_proj_w + b)
    k_dt<<<((size_t)MROWS * D_INNER + 255) / 256, 256, 0, stream>>>(xdbl, dt_proj_w,
                                                                    dt_proj_b, dt);
    // selective scan (128 B dynamic LDS for the async B_t/C_t broadcast)
    k_scan<<<BATCH, 256, 2 * D_STATE * sizeof(float), stream>>>(u_f32, dt, xdbl,
                                                                A_log, Dvec, y);
    // layernorm -> padded bf16
    k_ln<<<MROWS / 8, 256, 0, stream>>>(y, ln_g, ln_b, yln_bf);
    // GEMM3: o = y_ln W_out
    k_gemm3<<<MTILES / 2, 128, 0, stream>>>(yln_bf, Bsw3, o_buf);
    // batchnorm
    k_bnstats<<<D_MODEL, 256, 0, stream>>>(o_buf, stats);
    k_bnapply<<<((size_t)BATCH * D_MODEL * SEQ + 255) / 256, 256, 0, stream>>>(
        o_buf, stats, bn_g, bn_b, out);
}
